// BahdanauAttention_45208825758292
// MI455X (gfx1250) — matrix operand order repaired and checked by reading, not compiled, not run on hardware
//
#include <hip/hip_runtime.h>
#include <hip/hip_bf16.h>

// Problem constants (from reference)
#define B_  32
#define S_  2048
#define E_  1024
#define D_  1024
#define A_  512

typedef __attribute__((ext_vector_type(16))) __bf16 v16bf;
typedef __attribute__((ext_vector_type(8)))  __bf16 v8bf;
typedef __attribute__((ext_vector_type(8)))  float  v8f;
typedef __attribute__((ext_vector_type(4)))  int    v4i;
typedef __attribute__((address_space(1)))    v4i    gas_v4i;  // global v4i
typedef __attribute__((address_space(3)))    v4i    las_v4i;  // LDS v4i

#if __has_builtin(__builtin_amdgcn_tanhf)
#define TANH_F32(x) __builtin_amdgcn_tanhf(x)   // native v_tanh_f32 on gfx1250
#else
#define TANH_F32(x) tanhf(x)
#endif

#if __has_builtin(__builtin_amdgcn_global_load_async_to_lds_b128)
#define HAVE_ASYNC_LDS 1
#else
#define HAVE_ASYNC_LDS 0
#endif

// ---------------------------------------------------------------------------
// Prep 1: We [E,A] fp32  ->  Wt [A,E] bf16 (column-of-We contiguous in K).
// ---------------------------------------------------------------------------
__global__ void prep_wt_kernel(const float* __restrict__ We,
                               __bf16* __restrict__ Wt) {
    const int a = blockIdx.x;                     // 0..511
    for (int e = threadIdx.x; e < E_; e += 256)
        Wt[(size_t)a * E_ + e] = (__bf16)We[(size_t)e * A_ + a];
}

// ---------------------------------------------------------------------------
// Prep 2: dec_proj[b][a] = decoder_hidden[b,:] @ Wd[:,a] + bd[a] + be[a]
// ---------------------------------------------------------------------------
__global__ void prep_decproj_kernel(const float* __restrict__ dec,
                                    const float* __restrict__ Wd,
                                    const float* __restrict__ bd,
                                    const float* __restrict__ be,
                                    float* __restrict__ decp) {
    const int idx = blockIdx.x * 256 + threadIdx.x;   // 0..16383
    const int b = idx >> 9;
    const int a = idx & (A_ - 1);
    float acc = 0.f;
    const float* dh = dec + (size_t)b * D_;
    for (int e = 0; e < D_; ++e)
        acc += dh[e] * Wd[(size_t)e * A_ + a];
    decp[idx] = acc + bd[a] + be[a];
}

// ---------------------------------------------------------------------------
// Fused: enc_e = enc@We ; energy = tanh(enc_e + dec_e) ; scores = energy.v_w
//
// Block = 128 threads (4 waves) owning 64 M-rows. enc tile (64x1024) is
// converted fp32->bf16 into LDS; the per-N-tile B slice (16x1024 bf16, 32 KB)
// is staged into LDS once per block via async global->LDS copies and shared
// by all 4 waves. XOR swizzle on 16B granules keeps ds_load_b128 bank-free.
// LDS = (64+16)*1024*2 = 160 KB -> 2 blocks / WGP (8 waves).
// Epilogue scalars (dec_proj, v_w) are software-pipelined across N-tiles so
// their global-load latency hides under the 32-WMMA K-loop.
// ---------------------------------------------------------------------------
__global__ void scores_wmma_kernel(const float* __restrict__ enc,
                                   const __bf16* __restrict__ Wt,
                                   const float* __restrict__ decp,
                                   const float* __restrict__ vw,
                                   const float* __restrict__ vb,
                                   float* __restrict__ scores) {
    extern __shared__ __bf16 lds[];
    __bf16* ldsA = lds;                 // 64 rows x 1024 K, swizzled granules
    __bf16* ldsB = lds + 64 * E_;       // 16 cols x 1024 K, swizzled granules
    const int tid  = threadIdx.x;
    const int lane = tid & 31;
    const int wave = tid >> 5;
    const long rowBase = (long)blockIdx.x * 64;   // first M row of this block
    const int  b = (int)(rowBase >> 11);          // 2048 rows per batch

    // ---- stage A: 64x1024 fp32 -> bf16, one 16B granule (8 elems) per iter
    {
        const float4* g4 = (const float4*)(enc + rowBase * E_);
        for (int g = tid; g < 64 * (E_ / 8); g += 128) {
            const int row = g >> 7;               // 128 granules per row
            const int cb  = g & 127;
            float4 v0 = g4[g * 2 + 0];
            float4 v1 = g4[g * 2 + 1];
            v8bf p;
            p[0] = (__bf16)v0.x; p[1] = (__bf16)v0.y;
            p[2] = (__bf16)v0.z; p[3] = (__bf16)v0.w;
            p[4] = (__bf16)v1.x; p[5] = (__bf16)v1.y;
            p[6] = (__bf16)v1.z; p[7] = (__bf16)v1.w;
            *(v8bf*)(ldsA + (size_t)row * E_ + (size_t)((cb ^ (row & 15)) << 3)) = p;
        }
    }

    // ISA 16-bit A layout: lane<16 -> M=lane, K-granules {k4+0, k4+2} (+hi);
// B layout (mirror):   lane<16 -> N=lane, K-granules {k4+0, k4+2};
//                      lane>=16 same N, K-granules {k4+1, k4+3}.
    const int nCol   = lane & 15;
    const int hiHalf = lane >> 4;
    const size_t aBase = (size_t)(wave * 16 + nCol) * E_;  // (row&15)==nCol
    const size_t bBase = (size_t)nCol * E_;

    float spart[8];
#pragma unroll
    for (int i = 0; i < 8; ++i) spart[i] = 0.f;

    // Pipelined epilogue scalars for nt = 0.
    float dpeCur = decp[b * A_ + nCol];
    float vwCur  = vw[nCol];

    for (int nt = 0; nt < A_ / 16; ++nt) {        // 32 N-tiles
        const int n0 = nt * 16;

        // ---- stage B tile: 16x1024 bf16 (32 KB), async global -> LDS ----
        const __bf16* bsrc = Wt + (size_t)n0 * E_;
        for (int g = tid; g < 16 * (E_ / 8); g += 128) {
            const int row = g >> 7;
            const int cb  = g & 127;
            __bf16* dst = ldsB + (size_t)row * E_ + (size_t)((cb ^ row) << 3);
#if HAVE_ASYNC_LDS
            __builtin_amdgcn_global_load_async_to_lds_b128(
                (gas_v4i*)(bsrc + (size_t)g * 8),
                (las_v4i*)dst, 0, 0);
#else
            *(v8bf*)dst = *(const v8bf*)(bsrc + (size_t)g * 8);
#endif
        }

        // Prefetch next tile's epilogue scalars: the loads issue here and
        // their latency is covered by the 32-WMMA K-loop below (index is
        // clamped instead of branched so EXEC stays untouched).
        const int nn = (nt + 1 < A_ / 16) ? (nt + 1) * 16 : nt * 16;
        const float dpeNxt = decp[b * A_ + nn + nCol];
        const float vwNxt  = vw[nn + nCol];

#if HAVE_ASYNC_LDS
#if __has_builtin(__builtin_amdgcn_s_wait_asynccnt)
        __builtin_amdgcn_s_wait_asynccnt(0);
#else
        asm volatile("s_wait_asynccnt 0x0" ::: "memory");
#endif
#endif
        __syncthreads();

        v8f acc = {};
#pragma unroll 8
        for (int kt = 0; kt < E_ / 32; ++kt) {    // 32 K-steps
            const int k4 = kt * 4;                // 16B-granule index base
            v8bf alo = *(const v8bf*)(ldsA + aBase + (size_t)(((k4 + hiHalf)     ^ nCol) << 3));
            v8bf ahi = *(const v8bf*)(ldsA + aBase + (size_t)(((k4 + 2 + hiHalf) ^ nCol) << 3));
            v8bf blo = *(const v8bf*)(ldsB + bBase + (size_t)(((k4 + hiHalf)     ^ nCol) << 3));
            v8bf bhi = *(const v8bf*)(ldsB + bBase + (size_t)(((k4 + 2 + hiHalf) ^ nCol) << 3));
            v16bf av = __builtin_shufflevector(alo, ahi,
                0,1,2,3,4,5,6,7,8,9,10,11,12,13,14,15);
            v16bf bv = __builtin_shufflevector(blo, bhi,
                0,1,2,3,4,5,6,7,8,9,10,11,12,13,14,15);
            acc = __builtin_amdgcn_wmma_f32_16x16x32_bf16(
                false, av, false, bv, (short)0, acc, false, false);
        }

        // Fused epilogue on the C tile: lane holds column a = n0 + nCol,
        // VGPR i holds row (i + 8*hiHalf). Native v_tanh_f32.
#pragma unroll
        for (int i = 0; i < 8; ++i)
            spart[i] += vwCur * TANH_F32(acc[i] + dpeCur);
        dpeCur = dpeNxt;
        vwCur  = vwNxt;

        __syncthreads();   // before next iteration overwrites ldsB
    }

    // Reduce partial scores across each 16-lane half (columns of the tile).
#pragma unroll
    for (int i = 0; i < 8; ++i) {
        float s = spart[i];
        s += __shfl_xor(s, 1, 32);
        s += __shfl_xor(s, 2, 32);
        s += __shfl_xor(s, 4, 32);
        s += __shfl_xor(s, 8, 32);
        spart[i] = s;
    }
    const float vbv = vb[0];
    const long r0 = rowBase + wave * 16;
    if (lane == 0) {
#pragma unroll
        for (int i = 0; i < 8; ++i) scores[r0 + i] = spart[i] + vbv;
    } else if (lane == 16) {
#pragma unroll
        for (int i = 0; i < 8; ++i) scores[r0 + 8 + i] = spart[i] + vbv;
    }
}

// ---------------------------------------------------------------------------
// Softmax over S=2048 per batch. One 256-thread block per batch.
// ---------------------------------------------------------------------------
__global__ void softmax_kernel(const float* __restrict__ scores,
                               float* __restrict__ attn) {
    __shared__ float red[256];
    const int b = blockIdx.x, tid = threadIdx.x;
    const float* sc = scores + (size_t)b * S_;
    float loc[8];
    float m = -1e30f;
#pragma unroll
    for (int i = 0; i < 8; ++i) { loc[i] = sc[tid + i * 256]; m = fmaxf(m, loc[i]); }
    red[tid] = m; __syncthreads();
    for (int o = 128; o > 0; o >>= 1) {
        if (tid < o) red[tid] = fmaxf(red[tid], red[tid + o]);
        __syncthreads();
    }
    m = red[0]; __syncthreads();
    float s = 0.f;
#pragma unroll
    for (int i = 0; i < 8; ++i) { loc[i] = __expf(loc[i] - m); s += loc[i]; }
    red[tid] = s; __syncthreads();
    for (int o = 128; o > 0; o >>= 1) {
        if (tid < o) red[tid] += red[tid + o];
        __syncthreads();
    }
    const float inv = 1.f / red[0];
    float* out = attn + (size_t)b * S_;
#pragma unroll
    for (int i = 0; i < 8; ++i) out[tid + i * 256] = loc[i] * inv;
}

// ---------------------------------------------------------------------------
// context[b,e] = sum_s attn[b,s] * enc[b,s,e] : bandwidth-bound; split-S
// partials (deterministic, no fp32 atomics) then reduce.
// ---------------------------------------------------------------------------
__global__ void ctx_partial_kernel(const float* __restrict__ enc,
                                   const float* __restrict__ attn,
                                   float* __restrict__ partial) {
    const int b = blockIdx.x, ch = blockIdx.y, tid = threadIdx.x;
    const float* at = attn + (size_t)b * S_ + ch * 128;
    const float4* e4 = (const float4*)(enc + ((size_t)b * S_ + ch * 128) * E_) + tid;
    float4 acc = {0.f, 0.f, 0.f, 0.f};
    for (int s = 0; s < 128; ++s) {
        const float a = at[s];
        float4 v = e4[(size_t)s * (E_ / 4)];
        acc.x += a * v.x; acc.y += a * v.y;
        acc.z += a * v.z; acc.w += a * v.w;
    }
    ((float4*)(partial + ((size_t)(b * 16 + ch)) * E_))[tid] = acc;
}

__global__ void ctx_reduce_kernel(const float* __restrict__ partial,
                                  float* __restrict__ ctx) {
    const int b = blockIdx.x;
    const int e = blockIdx.y * 256 + threadIdx.x;
    float s = 0.f;
    for (int c = 0; c < 16; ++c)
        s += partial[((size_t)(b * 16 + c)) * E_ + e];
    ctx[(size_t)b * E_ + e] = s;
}

// ---------------------------------------------------------------------------
extern "C" void kernel_launch(void* const* d_in, const int* in_sizes, int n_in,
                              void* d_out, int out_size, void* d_ws, size_t ws_size,
                              hipStream_t stream) {
    const float* enc = (const float*)d_in[0];   // [32,2048,1024]
    const float* dec = (const float*)d_in[1];   // [32,1024]
    const float* We  = (const float*)d_in[2];   // [1024,512]
    const float* be  = (const float*)d_in[3];   // [512]
    const float* Wd  = (const float*)d_in[4];   // [1024,512]
    const float* bd  = (const float*)d_in[5];   // [512]
    const float* vw  = (const float*)d_in[6];   // [512]
    const float* vb  = (const float*)d_in[7];   // scalar

    float* dout = (float*)d_out;                // context [32,1024] ++ attn [32,2048]
    float* ctx  = dout;
    float* attn = dout + B_ * E_;

    char* w = (char*)d_ws;
    __bf16* Wt     = (__bf16*)w;                                  // 1 MB
    float*  decp   = (float*)(w + (1u << 20));                    // 64 KB
    float*  scores = (float*)(w + (1u << 20) + 65536u);           // 256 KB
    float*  part   = (float*)(w + (1u << 20) + 65536u + 262144u); // 2 MB

    prep_wt_kernel<<<A_, 256, 0, stream>>>(We, Wt);
    prep_decproj_kernel<<<(B_ * A_) / 256, 256, 0, stream>>>(dec, Wd, bd, be, decp);

    // 65536 M-rows / 64 per block = 1024 blocks; 160 KB dynamic LDS each.
    scores_wmma_kernel<<<(B_ * S_) / 64, 128,
                         (64 + 16) * E_ * sizeof(__bf16), stream>>>(
        enc, Wt, decp, vw, vb, scores);

    softmax_kernel<<<B_, 256, 0, stream>>>(scores, attn);

    ctx_partial_kernel<<<dim3(B_, 16), 256, 0, stream>>>(enc, attn, part);
    ctx_reduce_kernel<<<dim3(B_, E_ / 256), 256, 0, stream>>>(part, ctx);
}
